// MLayer_69020124446824
// MI455X (gfx1250) — matrix operand order repaired and checked by reading, not compiled, not run on hardware
//
#include <hip/hip_runtime.h>
#include <hip/hip_bf16.h>
#include <math.h>

// ---- problem constants (match reference) ----
#define DIMD 1024
#define HH   4
#define HDD  256
#define RR   16
#define NTOK 16384                  // B*S = 4*4096
#define EPSV 1e-5f
#define DSCALE 0.4082482904638631f  // 1/sqrt(6)

typedef __attribute__((ext_vector_type(16))) __bf16 v16bf;
typedef __attribute__((ext_vector_type(8)))  __bf16 v8bf;
typedef __attribute__((ext_vector_type(8)))  float  v8f;
typedef __attribute__((ext_vector_type(4)))  unsigned int u32x4;
typedef __attribute__((ext_vector_type(8)))  int i32x8;
typedef __attribute__((ext_vector_type(4)))  int i32x4;

#if __has_builtin(__builtin_amdgcn_tensor_load_to_lds)
#define HAVE_TDM 1
#endif

// ---------------------------------------------------------------------------
// helpers
// ---------------------------------------------------------------------------
__device__ inline float wave_sum(float v) {
#pragma unroll
  for (int o = 16; o > 0; o >>= 1) v += __shfl_xor(v, o, 32);
  return v;
}

__device__ inline float block_reduce(float val, float* sbuf) {
  int tid = threadIdx.x;
  sbuf[tid] = val;
  __syncthreads();
#pragma unroll
  for (int s = 64; s > 0; s >>= 1) {
    if (tid < s) sbuf[tid] += sbuf[tid + s];
    __syncthreads();
  }
  float r = sbuf[0];
  __syncthreads();
  return r;
}

__device__ inline float sigmoidf(float z) { return 1.0f / (1.0f + __expf(-z)); }

// A-fragment (16x32 MxK, bf16, row-major A with leading dim lda).
// ISA layout: lanes 0-15 -> M=lane, K chunks {8h..8h+7, 16+8h..16+8h+7}, h=lane>>4.
__device__ inline v16bf load_a_frag(const __bf16* __restrict__ A, int lda,
                                    int kb, int lane) {
  int m = lane & 15, half = lane >> 4;
  const __bf16* p = A + (size_t)m * lda + kb + 8 * half;
  v8bf lo = *(const v8bf*)p;
  v8bf hi = *(const v8bf*)(p + 16);
  v16bf a;
#pragma unroll
  for (int i = 0; i < 8; i++) { a[i] = lo[i]; a[i + 8] = hi[i]; }
  return a;
}

// B-fragment (32x16 KxN, bf16) loaded from B^T stored row-major [N][K] (ldb=K).
// ISA layout: n = lane&15; K chunks {8h..8h+7, 16+8h..16+8h+7}, h=lane>>4.
__device__ inline v16bf load_b_frag(const __bf16* __restrict__ Bt, int ldb,
                                    int kb, int lane) {
  int n = lane & 15, half = lane >> 4;
  const __bf16* p = Bt + (size_t)n * ldb + kb + 8 * half;
  v8bf lo = *(const v8bf*)p;
  v8bf hi = *(const v8bf*)(p + 16);
  v16bf b;
#pragma unroll
  for (int i = 0; i < 8; i++) { b[i] = lo[i]; b[i + 8] = hi[i]; }
  return b;
}

__device__ inline v8f wmma_bf16(v16bf a, v16bf b, v8f c) {
  return __builtin_amdgcn_wmma_f32_16x16x32_bf16(false, a, false, b,
                                                 (short)0, c, false, false);
}

// ---------------------------------------------------------------------------
// Tensor Data Mover: stage a contiguous [rows x rowlen] bf16 tile into LDS.
// D# per CDNA5 ISA ch.8: group0 = {count/flags, lds_addr, global_addr, type},
// group1 = {data_size, tensor/tile dims, dim0 stride}. 2-D tile.
// ---------------------------------------------------------------------------
__device__ inline void tdm_load_tile_bf16(unsigned lds_addr,
                                          const __bf16* gptr,
                                          unsigned rows, unsigned rowlen) {
#ifdef HAVE_TDM
  unsigned long long ga = (unsigned long long)(uintptr_t)gptr;
  u32x4 g0;
  g0.x = 1u;                                                  // count=1 (valid)
  g0.y = lds_addr;                                            // LDS byte addr
  g0.z = (unsigned)ga;                                        // global_addr lo
  g0.w = (unsigned)((ga >> 32) & 0x1FFFFFFu) | (2u << 30);    // addr hi | type=2
  i32x8 g1;
  g1[0] = (int)(1u << 16);                                    // data_size=1 (2B)
  g1[1] = (int)((rowlen & 0xFFFFu) << 16);                    // tensor_dim0 lo16
  g1[2] = (int)((rowlen >> 16) | ((rows & 0xFFFFu) << 16));   // dim0 hi | dim1 lo
  g1[3] = (int)(((rows >> 16) & 0xFFFFu) | ((rowlen & 0xFFFFu) << 16)); // tile_dim0
  g1[4] = (int)(rows & 0xFFFFu);                              // tile_dim1, dim2=0
  g1[5] = (int)rowlen;                                        // dim0_stride lo32
  g1[6] = 0;                                                  // stride hi / dim1 stride
  g1[7] = 0;
  i32x4 gz = {0, 0, 0, 0};
#if __clang_major__ >= 23
  i32x8 gz8 = {0, 0, 0, 0, 0, 0, 0, 0};
  __builtin_amdgcn_tensor_load_to_lds(g0, g1, gz, gz, gz8, 0);
#else
  __builtin_amdgcn_tensor_load_to_lds(g0, g1, gz, gz, 0);
#endif
#if __has_builtin(__builtin_amdgcn_s_wait_tensorcnt)
  __builtin_amdgcn_s_wait_tensorcnt(0);
#endif
#endif
}

// ---------------------------------------------------------------------------
// weight conversion: Wo [K=1024][N=1024] f32 -> Wt [N][K] bf16
// ---------------------------------------------------------------------------
__global__ void k_conv_wo(const float* __restrict__ W, __bf16* __restrict__ Wt) {
  int idx = blockIdx.x * blockDim.x + threadIdx.x;   // 0 .. 1M-1
  int k = idx >> 10, n = idx & 1023;
  Wt[(size_t)n * DIMD + k] = (__bf16)W[idx];
}

// fric_w [H][C=512][D=256] f32 -> fricT [H][D][C] bf16
__global__ void k_conv_fric(const float* __restrict__ W, __bf16* __restrict__ Wt) {
  int idx = blockIdx.x * blockDim.x + threadIdx.x;   // 0 .. 524287
  int h = idx / (2 * HDD * HDD);
  int rem = idx - h * (2 * HDD * HDD);
  int c = rem / HDD;
  int d = rem - c * HDD;
  Wt[(size_t)h * (2 * HDD * HDD) + (size_t)d * (2 * HDD) + c] = (__bf16)W[idx];
}

// ---------------------------------------------------------------------------
// LayerNorm -> bf16 xn, vn.  One 128-thread block per token.
// ---------------------------------------------------------------------------
__global__ void k_ln(const float* __restrict__ x, const float* __restrict__ v,
                     const float* __restrict__ gx, const float* __restrict__ bx,
                     const float* __restrict__ gv, const float* __restrict__ bv,
                     __bf16* __restrict__ xn, __bf16* __restrict__ vn) {
  __shared__ float sbuf[128];
  int t = blockIdx.x, tid = threadIdx.x;
  const float* xr = x + (size_t)t * DIMD;
  const float* vr = v + (size_t)t * DIMD;
  float xs[8], vs[8], sx = 0, sxx = 0, sv = 0, svv = 0;
#pragma unroll
  for (int j = 0; j < 8; j++) {
    int i = tid * 8 + j;
    xs[j] = xr[i]; vs[j] = vr[i];
    sx += xs[j]; sxx += xs[j] * xs[j];
    sv += vs[j]; svv += vs[j] * vs[j];
  }
  float tsx  = block_reduce(sx,  sbuf);
  float tsxx = block_reduce(sxx, sbuf);
  float tsv  = block_reduce(sv,  sbuf);
  float tsvv = block_reduce(svv, sbuf);
  float mux = tsx * (1.0f / DIMD), varx = tsxx * (1.0f / DIMD) - mux * mux;
  float muv = tsv * (1.0f / DIMD), varv = tsvv * (1.0f / DIMD) - muv * muv;
  float rx = rsqrtf(varx + EPSV), rv = rsqrtf(varv + EPSV);
  v8bf ox, ov;
#pragma unroll
  for (int j = 0; j < 8; j++) {
    int i = tid * 8 + j;
    ox[j] = (__bf16)((xs[j] - mux) * rx * gx[i] + bx[i]);
    ov[j] = (__bf16)((vs[j] - muv) * rv * gv[i] + bv[i]);
  }
  *(v8bf*)(xn + (size_t)t * DIMD + tid * 8) = ox;
  *(v8bf*)(vn + (size_t)t * DIMD + tid * 8) = ov;
}

// ---------------------------------------------------------------------------
// gamma GEMM: gammaPre[t, h*256+d] = sum_c cat[t,h,c] * fric_w[h,c,d]
// per head: M=16384, N=256, K=512 (K 0..255 from xn slice, 256..511 from vn).
// block = 128 threads = 4 waves, wave -> one 16x16 tile; grid (16, 256, H)
// ---------------------------------------------------------------------------
__global__ void k_gamma_gemm(const __bf16* __restrict__ xn,
                             const __bf16* __restrict__ vn,
                             const __bf16* __restrict__ fricT,
                             float* __restrict__ gammaPre) {
  int h = blockIdx.z;
  int wave = threadIdx.x >> 5, lane = threadIdx.x & 31;
  int n0 = blockIdx.x * 16;                 // d-tile within head
  int m0 = (blockIdx.y * 4 + wave) * 16;    // token tile
  const __bf16* Bt = fricT + (size_t)h * (2 * HDD * HDD);  // [d][c], ldb=512
  v8f acc = {};
#pragma unroll
  for (int kb = 0; kb < 2 * HDD; kb += 32) {
    const __bf16* Asrc = (kb < HDD)
        ? (xn + (size_t)h * HDD + kb)
        : (vn + (size_t)h * HDD + (kb - HDD));
    v16bf a = load_a_frag(Asrc + (size_t)m0 * DIMD, DIMD, 0, lane);
    v16bf b = load_b_frag(Bt + (size_t)n0 * (2 * HDD), 2 * HDD, kb, lane);
    acc = wmma_bf16(a, b, acc);
  }
  int half = lane >> 4, n = n0 + (lane & 15);
#pragma unroll
  for (int r = 0; r < 8; r++) {
    int m = m0 + r + 8 * half;
    gammaPre[(size_t)m * DIMD + h * HDD + n] = acc[r];
  }
}

// ---------------------------------------------------------------------------
// Heun (RK2) geodesic step.  One 128-thread block per token, wave h = head.
// Lane owns 8 contiguous d's of its head.
// ---------------------------------------------------------------------------
__device__ inline void acc_fn(const float xq[8], const float vq[8],
                              const float Fh[8], const float gm[8],
                              const float* __restrict__ Uh,
                              const float* __restrict__ Wxh,
                              const float* __restrict__ Vph,
                              int lane, float out[8]) {
  float s[RR];
#pragma unroll
  for (int r = 0; r < RR; r++) {
    const float* Ur = Uh  + r * HDD + lane * 8;
    const float* Wr = Wxh + r * HDD + lane * 8;
    float pc = 0.0f, pg = 0.0f;
#pragma unroll
    for (int j = 0; j < 8; j++) { pc += vq[j] * Ur[j]; pg += xq[j] * Wr[j]; }
    pc = wave_sum(pc);
    pg = wave_sum(pg);
    float g = tanhf(pg);
    s[r] = pc * pc * g;
  }
#pragma unroll
  for (int j = 0; j < 8; j++) {
    const float* Vrow = Vph + (size_t)(lane * 8 + j) * RR;
    float gv = 0.0f;
#pragma unroll
    for (int r = 0; r < RR; r++) gv += s[r] * Vrow[r];
    out[j] = -gv + Fh[j] - gm[j] * vq[j];
  }
}

__global__ void k_heun(const float* __restrict__ x, const float* __restrict__ v,
                       const float* __restrict__ F,
                       const float* __restrict__ gx, const float* __restrict__ bx,
                       const float* __restrict__ gv, const float* __restrict__ bv,
                       const float* __restrict__ U, const float* __restrict__ Wx,
                       const float* __restrict__ Vp,
                       const float* __restrict__ gate_w,
                       const float* __restrict__ gate_b,
                       const float* __restrict__ fric_b,
                       const float* __restrict__ dtp,
                       const float* __restrict__ gammaPre,
                       __bf16* __restrict__ xnew, __bf16* __restrict__ vnew) {
  __shared__ float sbuf[128];
  int t = blockIdx.x, tid = threadIdx.x;
  int h = tid >> 5, lane = tid & 31;

  // f32 LayerNorm (recomputed; exact)
  const float* xr = x + (size_t)t * DIMD;
  const float* vr = v + (size_t)t * DIMD;
  float xh[8], vh[8], sx = 0, sxx = 0, sv = 0, svv = 0;
#pragma unroll
  for (int j = 0; j < 8; j++) {
    int i = tid * 8 + j;
    xh[j] = xr[i]; vh[j] = vr[i];
    sx += xh[j]; sxx += xh[j] * xh[j];
    sv += vh[j]; svv += vh[j] * vh[j];
  }
  float tsx  = block_reduce(sx,  sbuf);
  float tsxx = block_reduce(sxx, sbuf);
  float tsv  = block_reduce(sv,  sbuf);
  float tsvv = block_reduce(svv, sbuf);
  float mux = tsx * (1.0f / DIMD), varx = tsxx * (1.0f / DIMD) - mux * mux;
  float muv = tsv * (1.0f / DIMD), varv = tsvv * (1.0f / DIMD) - muv * muv;
  float rx = rsqrtf(varx + EPSV), rv = rsqrtf(varv + EPSV);
#pragma unroll
  for (int j = 0; j < 8; j++) {
    int i = tid * 8 + j;
    xh[j] = (xh[j] - mux) * rx * gx[i] + bx[i];
    vh[j] = (vh[j] - muv) * rv * gv[i] + bv[i];
  }

  // per-token, per-head dt gate
  float zg = 0.0f;
#pragma unroll
  for (int j = 0; j < 8; j++) {
    int d = lane * 8 + j;
    zg += xh[j] * gate_w[h * (2 * HDD) + d] +
          vh[j] * gate_w[h * (2 * HDD) + HDD + d];
  }
  zg = wave_sum(zg) + gate_b[h];
  float gate = sigmoidf(zg);
  float p = dtp[h];
  float sp = (p > 20.0f) ? p : log1pf(__expf(p));
  float dtv = sp * 2.0f * gate;

  // friction gamma (sigmoid of GEMM output + bias) and force F
  float Fh[8], gm[8];
#pragma unroll
  for (int j = 0; j < 8; j++) {
    int i = tid * 8 + j;
    Fh[j] = F[(size_t)t * DIMD + i];
    gm[j] = sigmoidf(gammaPre[(size_t)t * DIMD + i] + fric_b[h * HDD + lane * 8 + j]);
  }

  const float* Uh  = U  + (size_t)h * RR * HDD;
  const float* Wxh = Wx + (size_t)h * RR * HDD;
  const float* Vph = Vp + (size_t)h * HDD * RR;

  float k1v[8];
  acc_fn(xh, vh, Fh, gm, Uh, Wxh, Vph, lane, k1v);
  float xe[8], ve[8];
#pragma unroll
  for (int j = 0; j < 8; j++) { xe[j] = xh[j] + dtv * vh[j]; ve[j] = vh[j] + dtv * k1v[j]; }
  float k2v[8];
  acc_fn(xe, ve, Fh, gm, Uh, Wxh, Vph, lane, k2v);

  v8bf ox, ov;
#pragma unroll
  for (int j = 0; j < 8; j++) {
    ox[j] = (__bf16)(xh[j] + 0.5f * dtv * (vh[j] + ve[j]));     // k1x+k2x = vh+ve
    ov[j] = (__bf16)(vh[j] + 0.5f * dtv * (k1v[j] + k2v[j]));
  }
  *(v8bf*)(xnew + (size_t)t * DIMD + tid * 8) = ox;
  *(v8bf*)(vnew + (size_t)t * DIMD + tid * 8) = ov;
}

// ---------------------------------------------------------------------------
// output GEMM: out = res + (1/sqrt(6)) * (A_bf16 @ Wo)   [M=16384,N=K=1024]
// Block: 128 threads = 4 waves, each wave a 16(M)x32(N) tile (2 accumulators,
// A-fragment reused).  B panel [32 x 1024] staged in LDS once per block via
// the Tensor Data Mover.  grid (32, 256, 2).
// ---------------------------------------------------------------------------
__global__ void k_out_gemm(const __bf16* __restrict__ Ax,
                           const __bf16* __restrict__ Av,
                           const __bf16* __restrict__ WoxT,
                           const __bf16* __restrict__ WovT,
                           const float* __restrict__ xres,
                           const float* __restrict__ vres,
                           float* __restrict__ out) {
  __shared__ __bf16 smemB[32 * DIMD];   // 64 KB B panel
  int which = blockIdx.z;
  const __bf16* A   = which ? Av    : Ax;
  const __bf16* Bt  = which ? WovT  : WoxT;
  const float*  res = which ? vres  : xres;
  float* o = out + (size_t)which * NTOK * DIMD;

  int wave = threadIdx.x >> 5, lane = threadIdx.x & 31;
  int n0 = blockIdx.x * 32;
  int m0 = (blockIdx.y * 4 + wave) * 16;

  // stage B panel (rows n0..n0+31 of Bt, full K) into LDS
#ifdef HAVE_TDM
  if (threadIdx.x < 32) {
    tdm_load_tile_bf16((unsigned)(uintptr_t)&smemB[0],
                       Bt + (size_t)n0 * DIMD, 32u, (unsigned)DIMD);
  }
#else
  for (int i = threadIdx.x; i < 32 * (DIMD / 8); i += 128) {
    ((v8bf*)smemB)[i] = ((const v8bf*)(Bt + (size_t)n0 * DIMD))[i];
  }
#endif
  __syncthreads();

  const __bf16* Arow = A + (size_t)m0 * DIMD;
  v8f acc0 = {};
  v8f acc1 = {};
#pragma unroll 4
  for (int kb = 0; kb < DIMD; kb += 32) {
    __builtin_prefetch(Arow + (size_t)(lane & 15) * DIMD + kb + 64, 0, 1);
    v16bf a  = load_a_frag(Arow, DIMD, kb, lane);
    v16bf b0 = load_b_frag(smemB, DIMD, kb, lane);            // N rows 0..15
    v16bf b1 = load_b_frag(smemB + 16 * DIMD, DIMD, kb, lane); // N rows 16..31
    acc0 = wmma_bf16(a, b0, acc0);
    acc1 = wmma_bf16(a, b1, acc1);
  }
  int half = lane >> 4;
#pragma unroll
  for (int r = 0; r < 8; r++) {
    int m = m0 + r + 8 * half;
    int nA = n0 + (lane & 15);
    int nB = nA + 16;
    o[(size_t)m * DIMD + nA] = res[(size_t)m * DIMD + nA] + DSCALE * acc0[r];
    o[(size_t)m * DIMD + nB] = res[(size_t)m * DIMD + nB] + DSCALE * acc1[r];
  }
}

// ---------------------------------------------------------------------------
// launch
// ---------------------------------------------------------------------------
extern "C" void kernel_launch(void* const* d_in, const int* in_sizes, int n_in,
                              void* d_out, int out_size, void* d_ws, size_t ws_size,
                              hipStream_t stream) {
  (void)in_sizes; (void)n_in; (void)out_size; (void)ws_size;
  const float* x        = (const float*)d_in[0];
  const float* v        = (const float*)d_in[1];
  const float* F        = (const float*)d_in[2];
  const float* ln_x_g   = (const float*)d_in[3];
  const float* ln_x_b   = (const float*)d_in[4];
  const float* ln_v_g   = (const float*)d_in[5];
  const float* ln_v_b   = (const float*)d_in[6];
  const float* U        = (const float*)d_in[7];
  const float* Wx       = (const float*)d_in[8];
  const float* Vp       = (const float*)d_in[9];
  const float* gate_w   = (const float*)d_in[10];
  const float* gate_b   = (const float*)d_in[11];
  const float* fric_w   = (const float*)d_in[12];
  const float* fric_b   = (const float*)d_in[13];
  const float* dt_params= (const float*)d_in[14];
  const float* Wo_x     = (const float*)d_in[15];
  const float* Wo_v     = (const float*)d_in[16];
  float* out = (float*)d_out;

  // workspace layout (bytes)
  char* ws = (char*)d_ws;
  const size_t actBf = (size_t)NTOK * DIMD * sizeof(__bf16);   // 32 MB
  __bf16* xn_bf   = (__bf16*)(ws);
  __bf16* vn_bf   = (__bf16*)(ws + actBf);
  __bf16* xnew_bf = (__bf16*)(ws + 2 * actBf);
  __bf16* vnew_bf = (__bf16*)(ws + 3 * actBf);
  float*  gammaPre= (float*)(ws + 4 * actBf);                  // 64 MB
  char*   wbase   = ws + 4 * actBf + (size_t)NTOK * DIMD * sizeof(float);
  __bf16* WoxT_bf = (__bf16*)(wbase);
  __bf16* WovT_bf = (__bf16*)(wbase + (size_t)DIMD * DIMD * sizeof(__bf16));
  __bf16* fricT_bf= (__bf16*)(wbase + 2 * (size_t)DIMD * DIMD * sizeof(__bf16));

  // 1. weight conversion
  k_conv_wo  <<<(DIMD * DIMD) / 256, 256, 0, stream>>>(Wo_x, WoxT_bf);
  k_conv_wo  <<<(DIMD * DIMD) / 256, 256, 0, stream>>>(Wo_v, WovT_bf);
  k_conv_fric<<<(HH * 2 * HDD * HDD) / 256, 256, 0, stream>>>(fric_w, fricT_bf);

  // 2. LayerNorm -> bf16 activations
  k_ln<<<NTOK, 128, 0, stream>>>(x, v, ln_x_g, ln_x_b, ln_v_g, ln_v_b, xn_bf, vn_bf);

  // 3. friction-gate GEMM (WMMA bf16)
  k_gamma_gemm<<<dim3(HDD / 16, NTOK / 64, HH), 128, 0, stream>>>(
      xn_bf, vn_bf, fricT_bf, gammaPre);

  // 4. Heun geodesic integration (f32 pointwise + wave reductions)
  k_heun<<<NTOK, 128, 0, stream>>>(x, v, F, ln_x_g, ln_x_b, ln_v_g, ln_v_b,
                                   U, Wx, Vp, gate_w, gate_b, fric_b, dt_params,
                                   gammaPre, xnew_bf, vnew_bf);

  // 5. output projections + residual (WMMA bf16 from LDS-staged B, f32 epilogue)
  k_out_gemm<<<dim3(DIMD / 32, NTOK / 64, 2), 128, 0, stream>>>(
      xnew_bf, vnew_bf, WoxT_bf, WovT_bf, x, v, out);
}
